// HierarchicalMemoryEngine_54752243090069
// MI455X (gfx1250) — matrix operand order repaired and checked by reading, not compiled, not run on hardware
//
#include <hip/hip_runtime.h>
#include <hip/hip_bf16.h>
#include <math.h>

#define B_ 8
#define T_ 2048
#define H_ 1024
#define M_ 256
#define NC_ 16
#define CL_ (T_ / NC_)   // 128 steps per scan chunk

typedef __attribute__((ext_vector_type(16))) __bf16          v16bf;
typedef __attribute__((ext_vector_type(8)))  float           v8f;
typedef __attribute__((ext_vector_type(8)))  unsigned short  u16x8;

union FragBF {
  v16bf v;
  u16x8 h[2];
};

__device__ __forceinline__ unsigned short f2bf(float f) {
  // round-to-nearest-even f32 -> bf16 bits
  unsigned int u = __float_as_uint(f);
  unsigned int r = u + 0x7fffu + ((u >> 16) & 1u);
  return (unsigned short)(r >> 16);
}

// ---------------------------------------------------------------------------
// rec[b,h] = g_sr * sum_m ( t0re[b,m]*W[m,h] + t0im[b,m]*W[M+m,h] )
// ---------------------------------------------------------------------------
__global__ __launch_bounds__(256) void rec_kernel(
    const float* __restrict__ t0re, const float* __restrict__ t0im,
    const float* __restrict__ W, const float* __restrict__ g_sr,
    float* __restrict__ rec) {
  int i = blockIdx.x * 256 + threadIdx.x;   // B*H threads
  int h = i % H_;
  int b = i / H_;
  float g = g_sr[0];
  float s = 0.f;
  for (int m = 0; m < M_; ++m) {
    s += t0re[b * M_ + m] * W[(size_t)m * H_ + h];
    s += t0im[b * M_ + m] * W[(size_t)(M_ + m) * H_ + h];
  }
  rec[i] = g * s;
}

// ---------------------------------------------------------------------------
// f32 -> bf16 conversion (elementwise)
// ---------------------------------------------------------------------------
__global__ __launch_bounds__(256) void cvt_bf16_kernel(
    const float* __restrict__ in, unsigned short* __restrict__ out, size_t n) {
  size_t i = (size_t)blockIdx.x * 256 + threadIdx.x;
  if (i < n) out[i] = f2bf(in[i]);
}

// out[b][c][r] = in[b][r][c]  (f32 -> bf16, batched transpose)
__global__ __launch_bounds__(256) void cvtT_bf16_kernel(
    const float* __restrict__ in, unsigned short* __restrict__ out,
    int rows, int cols) {
  size_t i = (size_t)blockIdx.x * 256 + threadIdx.x;   // B*rows*cols threads
  int r = (int)(i % rows);
  size_t q = i / rows;
  int c = (int)(q % cols);
  int b = (int)(q / cols);
  out[i] = f2bf(in[((size_t)b * rows + r) * cols + c]);
}

// ---------------------------------------------------------------------------
// Batched bf16 WMMA GEMM:  C[b] = A[b] (f32) x B[b]   with B supplied
// PRE-TRANSPOSED as BT (bf16, [N][K] row-major) so both global and LDS
// traffic for the B tile is fully vectorized (b128).
// Optional fusion: A = hs + g_bu*inc + g_td*td + bias[k]  (GEMM-1)
// Workgroup tile 128(T) x 128(N), BK = 32, 8 waves in 2x4 arrangement,
// each wave owns a 64x32 output strip = 4x2 WMMA tiles.
// Ping-pong double-buffered LDS, one barrier per K-chunk.
// ---------------------------------------------------------------------------
__global__ __launch_bounds__(256) void gemm_bf16_kernel(
    const float* __restrict__ A,
    const float* __restrict__ hs, const float* __restrict__ inc,
    const float* __restrict__ td, const float* __restrict__ bias,
    const float* __restrict__ g_bu, const float* __restrict__ g_td,
    const unsigned short* __restrict__ BT,
    float* __restrict__ C,
    int K, int N, int fuse) {
  const int b  = blockIdx.z;
  const int n0 = blockIdx.x * 128;
  const int t0 = blockIdx.y * 128;

  __shared__ unsigned short AL[2][128 * 40];  // [row][k], stride 40 (pad)
  __shared__ unsigned short BL[2][128 * 40];  // [n][k],   stride 40 (pad)

  const int tid   = threadIdx.x;
  const int lane  = tid & 31;
  const int wid   = tid >> 5;
  const int waveM = wid & 1;    // 2 strips of 64 rows
  const int waveN = wid >> 1;   // 4 strips of 32 cols
  const int lm    = lane & 15;
  const int lh    = lane >> 4;  // half selector

  const float gbu = fuse ? g_bu[0] : 0.f;
  const float gtd = fuse ? g_td[0] : 0.f;

  const size_t abase = (size_t)b * T_ * K + (size_t)t0 * K;
  const float* pA  = fuse ? nullptr : (A + abase);
  const float* pHS = fuse ? (hs + abase) : nullptr;
  const float* pIN = fuse ? (inc + abase) : nullptr;
  const float* pTD = fuse ? (td + abase) : nullptr;
  const float* pBI = fuse ? (bias + (size_t)b * K) : nullptr;
  const unsigned short* pBT = BT + (size_t)b * N * K;
  float* pC = C + (size_t)b * T_ * N + (size_t)t0 * N + n0;

  // ---- staging helpers ----------------------------------------------------
  auto stageA = [&](int k0, int s) {
    const int kg = tid & 7;        // 8 groups of 4 floats across K
    const int r0 = tid >> 3;       // 0..31
#pragma unroll
    for (int rr = 0; rr < 4; ++rr) {
      const int r = r0 + rr * 32;
      const size_t off = (size_t)r * K + k0 + kg * 4;
      float4 v;
      if (fuse) {
        float4 x = *reinterpret_cast<const float4*>(pHS + off);
        float4 y = *reinterpret_cast<const float4*>(pIN + off);
        float4 z = *reinterpret_cast<const float4*>(pTD + off);
        const int kb = k0 + kg * 4;
        v.x = x.x + gbu * y.x + gtd * z.x + pBI[kb + 0];
        v.y = x.y + gbu * y.y + gtd * z.y + pBI[kb + 1];
        v.z = x.z + gbu * y.z + gtd * z.z + pBI[kb + 2];
        v.w = x.w + gbu * y.w + gtd * z.w + pBI[kb + 3];
      } else {
        v = *reinterpret_cast<const float4*>(pA + off);
      }
      uint2 pk;
      pk.x = (unsigned)f2bf(v.x) | ((unsigned)f2bf(v.y) << 16);
      pk.y = (unsigned)f2bf(v.z) | ((unsigned)f2bf(v.w) << 16);
      *reinterpret_cast<uint2*>(&AL[s][r * 40 + kg * 4]) = pk;
    }
  };
  auto stageB = [&](int k0, int s) {
    // BT is [N][K] row-major: straight vectorized copy, no transpose needed
#pragma unroll
    for (int it = 0; it < 2; ++it) {
      const int slot = tid + it * 256;   // 512 slots = 128 n x 4 k-groups(8)
      const int kg = slot & 3;
      const int n  = slot >> 2;
      uint4 raw = *reinterpret_cast<const uint4*>(
          pBT + (size_t)(n0 + n) * K + k0 + kg * 8);
      *reinterpret_cast<uint4*>(&BL[s][n * 40 + kg * 8]) = raw;
    }
  };

  v8f acc[4][2];
#pragma unroll
  for (int i = 0; i < 4; ++i)
#pragma unroll
    for (int j = 0; j < 2; ++j)
#pragma unroll
      for (int e = 0; e < 8; ++e) acc[i][j][e] = 0.f;

  const int nkc = K >> 5;
  stageA(0, 0);
  stageB(0, 0);
  __syncthreads();

  for (int kc = 0; kc < nkc; ++kc) {
    const int cur = kc & 1;

    // 1) fragment loads from current buffer (ISA 7.12.2 layouts)
    FragBF af[4], bf2[2];
#pragma unroll
    for (int mt = 0; mt < 4; ++mt) {
      const unsigned short* ap = &AL[cur][(waveM * 64 + mt * 16 + lm) * 40];
      af[mt].h[0] = *reinterpret_cast<const u16x8*>(ap + lh * 8);        // K 0..7 / 8..15
      af[mt].h[1] = *reinterpret_cast<const u16x8*>(ap + 16 + lh * 8);   // K 16..23 / 24..31
    }
#pragma unroll
    for (int nt = 0; nt < 2; ++nt) {
      const unsigned short* bp =
          &BL[cur][(waveN * 32 + nt * 16 + lm) * 40];
      bf2[nt].h[0] = *reinterpret_cast<const u16x8*>(bp + lh * 8);       // K 0..7 / 8..15
      bf2[nt].h[1] = *reinterpret_cast<const u16x8*>(bp + 16 + lh * 8);  // K 16..23 / 24..31
    }

    // 2) stage next chunk into the other buffer (overlaps with WMMAs below)
    if (kc + 1 < nkc) {
      stageA((kc + 1) << 5, cur ^ 1);
      stageB((kc + 1) << 5, cur ^ 1);
    }

    // 3) 8 WMMAs per wave
#pragma unroll
    for (int mt = 0; mt < 4; ++mt)
#pragma unroll
      for (int nt = 0; nt < 2; ++nt)
        acc[mt][nt] = __builtin_amdgcn_wmma_f32_16x16x32_bf16(
            false, af[mt].v, false, bf2[nt].v, (short)0, acc[mt][nt], false, false);

    if (kc + 1 < nkc) __syncthreads();   // uniform condition
  }

  // ---- store C (f32): VGPR j -> row j + 8*(lane>=16), col lane&15 ----
#pragma unroll
  for (int mt = 0; mt < 4; ++mt)
#pragma unroll
    for (int nt = 0; nt < 2; ++nt) {
      const int col   = waveN * 32 + nt * 16 + lm;
      const int rbase = waveM * 64 + mt * 16 + lh * 8;
#pragma unroll
      for (int j = 0; j < 8; ++j)
        pC[(size_t)(rbase + j) * N + col] = acc[mt][nt][j];
    }
}

// ---------------------------------------------------------------------------
// Parallel linear scan: s_t = g*s_{t-1} + beta*a_t ; c_t = a_t * s_t
// 3-pass chunked scan over T (NC_ chunks of CL_ steps).
// ---------------------------------------------------------------------------
__device__ __forceinline__ void get_gb(const float* graw, const float* braw,
                                       float& g, float& bet) {
  g = 1.f / (1.f + __expf(-graw[0]));
  bet = log1pf(__expf(braw[0]));
}

__global__ __launch_bounds__(256) void scan_pass1(
    const float* __restrict__ alpha, const float* __restrict__ graw,
    const float* __restrict__ braw, float* __restrict__ carry) {
  const int m = threadIdx.x, c = blockIdx.x, b = blockIdx.y;
  float g, bet; get_gb(graw, braw, g, bet);
  const float* a = alpha + ((size_t)b * T_ + (size_t)c * CL_) * M_ + m;
  float s = 0.f;
#pragma unroll 4
  for (int i = 0; i < CL_; ++i) s = g * s + bet * a[(size_t)i * M_];
  carry[((size_t)b * NC_ + c) * M_ + m] = s;
}

__global__ __launch_bounds__(256) void scan_pass2(
    const float* __restrict__ carry, const float* __restrict__ t0re,
    const float* __restrict__ graw, const float* __restrict__ braw,
    float* __restrict__ carry_in) {
  const int m = threadIdx.x, b = blockIdx.x;
  float g, bet; get_gb(graw, braw, g, bet);
  const float gp = __powf(g, (float)CL_);
  float x = t0re[b * M_ + m];
  for (int c = 0; c < NC_; ++c) {
    const size_t idx = ((size_t)b * NC_ + c) * M_ + m;
    carry_in[idx] = x;
    x = gp * x + carry[idx];
  }
}

__global__ __launch_bounds__(256) void scan_pass3(
    const float* __restrict__ alpha, const float* __restrict__ carry_in,
    const float* __restrict__ graw, const float* __restrict__ braw,
    float* __restrict__ cre) {
  const int m = threadIdx.x, c = blockIdx.x, b = blockIdx.y;
  float g, bet; get_gb(graw, braw, g, bet);
  const size_t base = ((size_t)b * T_ + (size_t)c * CL_) * M_ + m;
  float s = carry_in[((size_t)b * NC_ + c) * M_ + m];
#pragma unroll 4
  for (int i = 0; i < CL_; ++i) {
    const float a = alpha[base + (size_t)i * M_];
    s = g * s + bet * a;
    cre[base + (size_t)i * M_] = a * s;
  }
}

// ---------------------------------------------------------------------------
extern "C" void kernel_launch(void* const* d_in, const int* in_sizes, int n_in,
                              void* d_out, int out_size, void* d_ws, size_t ws_size,
                              hipStream_t stream) {
  const float* hs    = (const float*)d_in[0];
  const float* inc   = (const float*)d_in[1];
  const float* td    = (const float*)d_in[2];
  const float* basis = (const float*)d_in[3];
  const float* coup  = (const float*)d_in[4];
  const float* Wrec  = (const float*)d_in[5];
  const float* t0re  = (const float*)d_in[6];
  const float* t0im  = (const float*)d_in[7];
  const float* g_bu  = (const float*)d_in[8];
  const float* g_td  = (const float*)d_in[9];
  const float* g_sr  = (const float*)d_in[10];
  const float* graw  = (const float*)d_in[11];
  const float* braw  = (const float*)d_in[12];

  char* ws = (char*)d_ws;
  float* rec              = (float*)ws;          ws += (size_t)B_ * H_ * 4;
  unsigned short* bas_bf  = (unsigned short*)ws; ws += (size_t)B_ * H_ * M_ * 2;  // basis   [H][M]
  unsigned short* basT_bf = (unsigned short*)ws; ws += (size_t)B_ * M_ * H_ * 2;  // basis^T [M][H]
  unsigned short* couT_bf = (unsigned short*)ws; ws += (size_t)B_ * M_ * M_ * 2;  // coupling^T [N][K]
  float* wbuf             = (float*)ws;          ws += (size_t)B_ * T_ * M_ * 4;  // w, reused as c_re
  float* alpha            = (float*)ws;          ws += (size_t)B_ * T_ * M_ * 4;
  float* carry            = (float*)ws;          ws += (size_t)B_ * NC_ * M_ * 4;
  float* carry_in         = (float*)ws;

  // small prep
  rec_kernel<<<(B_ * H_) / 256, 256, 0, stream>>>(t0re, t0im, Wrec, g_sr, rec);
  cvt_bf16_kernel<<<(B_ * H_ * M_) / 256, 256, 0, stream>>>(basis, bas_bf, (size_t)B_ * H_ * M_);
  cvtT_bf16_kernel<<<(B_ * M_ * H_) / 256, 256, 0, stream>>>(basis, basT_bf, H_, M_);
  cvtT_bf16_kernel<<<(B_ * M_ * M_) / 256, 256, 0, stream>>>(coup, couT_bf, M_, M_);

  // GEMM 1 (fused combined): w = combined @ basis      [T,1024]x[1024,256]
  //   B = basis [K=H,N=M]  ->  BT layout [N][K] = basis^T = basT_bf
  gemm_bf16_kernel<<<dim3(M_ / 128, T_ / 128, B_), 256, 0, stream>>>(
      nullptr, hs, inc, td, rec, g_bu, g_td, basT_bf, wbuf, H_, M_, 1);

  // GEMM 2: alpha = w @ coupling                       [T,256]x[256,256]
  //   B = coupling [K=M,N=M] -> BT layout = coupling^T = couT_bf
  gemm_bf16_kernel<<<dim3(M_ / 128, T_ / 128, B_), 256, 0, stream>>>(
      wbuf, nullptr, nullptr, nullptr, nullptr, nullptr, nullptr,
      couT_bf, alpha, M_, M_, 0);

  // scan (3-pass)
  scan_pass1<<<dim3(NC_, B_), 256, 0, stream>>>(alpha, graw, braw, carry);
  scan_pass2<<<B_, 256, 0, stream>>>(carry, t0re, graw, braw, carry_in);
  scan_pass3<<<dim3(NC_, B_), 256, 0, stream>>>(alpha, carry_in, graw, braw, wbuf);

  // GEMM 3: y = c_re @ basis^T                         [T,256]x[256,1024]
  //   B = basis^T [K=M,N=H] -> BT layout [N=H][K=M] = basis itself = bas_bf
  gemm_bf16_kernel<<<dim3(H_ / 128, T_ / 128, B_), 256, 0, stream>>>(
      wbuf, nullptr, nullptr, nullptr, nullptr, nullptr, nullptr,
      bas_bf, (float*)d_out, M_, H_, 0);
}
